// KNFP_GCN_76922864271373
// MI455X (gfx1250) — matrix operand order repaired and checked by reading, not compiled, not run on hardware
//
#include <hip/hip_runtime.h>
#include <hip/hip_bf16.h>
#include <math.h>

typedef __attribute__((ext_vector_type(16))) _Float16 v16h;
typedef __attribute__((ext_vector_type(8)))  _Float16 v8h;
typedef __attribute__((ext_vector_type(4)))  _Float16 v4h;
typedef __attribute__((ext_vector_type(8)))  float    v8f;
typedef __attribute__((ext_vector_type(4)))  float    v4f;
typedef __attribute__((ext_vector_type(2)))  unsigned int v2u;
typedef __attribute__((ext_vector_type(4)))  unsigned int v4u;
typedef __attribute__((ext_vector_type(4)))  int      v4i;
typedef __attribute__((ext_vector_type(8)))  int      v8i;

#define NNODES 102400
#define NEDGES 1600000
#define NGRAPH 256
#define HDIM   128
#define GHID   120
#define TSTEPS 19
#define XCW    4816

#if __has_builtin(__builtin_amdgcn_tensor_load_to_lds) && __has_builtin(__builtin_amdgcn_s_wait_tensorcnt)
#define HAVE_TDM 1
#endif

// ---------------- utility ----------------
__global__ void k_zero_f32(float* __restrict__ p, int n) {
  int i = blockIdx.x * blockDim.x + threadIdx.x;
  if (i < n) p[i] = 0.0f;
}

// ---------------- layer 1 (F_IN=4) ----------------
__global__ void k_edge_agg4(const float* __restrict__ x, const int* __restrict__ ei,
                            float* __restrict__ agg, int E) {
  int e = blockIdx.x * blockDim.x + threadIdx.x;
  if (e >= E) return;
  int s = ei[e], d = ei[E + e];
  v4f v = *(const v4f*)(x + (size_t)s * 4);
  float* ap = agg + (size_t)d * 4;
  #pragma unroll
  for (int i = 0; i < 4; ++i) atomicAdd(ap + i, v[i]);
}

__global__ void k_layer1(const float* __restrict__ agg, const float* __restrict__ x,
                         const float* __restrict__ Wrel, const float* __restrict__ Wroot,
                         const float* __restrict__ b, _Float16* __restrict__ out) {
  int idx = blockIdx.x * blockDim.x + threadIdx.x;
  if (idx >= NNODES * HDIM) return;
  int n = idx >> 7, h = idx & 127;
  float acc = b[h];
  #pragma unroll
  for (int k = 0; k < 4; ++k)
    acc += agg[(size_t)n * 4 + k] * Wrel[k * HDIM + h] + x[(size_t)n * 4 + k] * Wroot[k * HDIM + h];
  out[idx] = (_Float16)fmaxf(acc, 0.0f);
}

// ---- edge scatter-add, H=128: f16 source -> f16 agg via GLOBAL_ATOMIC_PK_ADD_F16 ----
// One wave per edge; each lane handles 4 features = 2 packed-f16 atomics.
__global__ void k_edge_agg128(const _Float16* __restrict__ xin, const int* __restrict__ ei,
                              _Float16* __restrict__ agg, int E) {
  int gid = blockIdx.x * blockDim.x + threadIdx.x;
  int e = gid >> 5;
  int lane = gid & 31;
  if (e >= E) return;
  int s = ei[e], d = ei[E + e];
  v2u v = *(const v2u*)(xin + (size_t)s * HDIM + lane * 4);
  unsigned long long addr = (unsigned long long)(agg + (size_t)d * HDIM + lane * 4);
  asm volatile("global_atomic_pk_add_f16 %0, %1, off"          :: "v"(addr), "v"(v[0]) : "memory");
  asm volatile("global_atomic_pk_add_f16 %0, %1, off offset:4" :: "v"(addr), "v"(v[1]) : "memory");
}

// ---------------- weight pre-swizzle for WMMA B fragments ----------------
// Wcat[k,c] (k<128: W_rel, k>=128: W_root), stored so each lane's 16 B values
// for tile (kt, ct) are contiguous: swz[((kt*8+ct)*32 + lane)*16 + i]
__global__ void k_prep_wswz(const float* __restrict__ Wrel, const float* __restrict__ Wroot,
                            _Float16* __restrict__ swz) {
  int idx = blockIdx.x * blockDim.x + threadIdx.x;
  if (idx >= 256 * HDIM) return;
  int k = idx >> 7, c = idx & 127;
  float w = (k < HDIM) ? Wrel[k * HDIM + c] : Wroot[(k - HDIM) * HDIM + c];
  int kt = k >> 5, ct = c >> 4;
  int kin = k & 31, lh = kin >> 4, i = kin & 15;
  int lane = (c & 15) + lh * 16;
  swz[(size_t)(((kt * 8 + ct) * 32) + lane) * 16 + i] = (_Float16)w;
}

#ifdef HAVE_TDM
// Issue one TDM 2-D tile load: tensor = [NNODES x 128] f16 row-major,
// tile = 16 rows x 128 cols starting at gaddr, into LDS offset lds_off.
__device__ __forceinline__ void tdm_load_tile(const _Float16* gaddr, unsigned lds_off) {
  unsigned long long ga = (unsigned long long)gaddr;
  // D# group0: [1:0]=count=1 | lds_addr | global_addr[56:0] | type=2 in [127:126]
  v4u g0 = { 1u,
             lds_off,
             (unsigned)ga,
             (unsigned)((ga >> 32) & 0x1FFFFFFu) | (2u << 30) };
  // D# group1: data_size=1(2B) @[17:16]; tensor_dim0=128 @[79:48];
  // tensor_dim1=NNODES @[111:80]; tile_dim0=128 @[127:112]; tile_dim1=16 @[143:128];
  // tensor_dim0_stride=128 @[207:160]
  v8i g1 = { (int)(1u << 16),
             (int)(128u << 16),
             (int)((NNODES & 0xFFFFu) << 16),
             (int)((NNODES >> 16) | (128u << 16)),
             16,
             128,
             0, 0 };
  v4i gz  = { 0, 0, 0, 0 };
  v8i gz8 = { 0, 0, 0, 0, 0, 0, 0, 0 };
  // clang-23 / therock-10.0 form: 6 args (g0, g1, g2, g3, extra, cpol)
  __builtin_amdgcn_tensor_load_to_lds(g0, g1, gz, gz, gz8, 0);
}
#endif

// ---------------- fused GraphConv GEMM via WMMA ----------------
// out = relu([agg_f16 | x_f16](M x 256) @ Wswz(256 x 128) + bias) [+ resid]
// block = 256 threads = 8 waves; wave w -> 16-col tile w; block -> 16 rows.
// A tile (16 x 256 f16 = 8KB) staged into LDS once per block via TDM, consumed
// with explicit ds_load_b128 (opaque to the compiler, matching the TDM write).
__global__ void __launch_bounds__(256)
k_gemm_gconv(const _Float16* __restrict__ agg, const _Float16* __restrict__ xin,
             const _Float16* __restrict__ wswz, const float* __restrict__ bias,
             const _Float16* __restrict__ resid, _Float16* __restrict__ out) {
  __shared__ __attribute__((aligned(16))) _Float16 Atile[2 * 16 * HDIM];  // [agg | x] @ LDS 0
  int lane = threadIdx.x & 31;
  int wave = threadIdx.x >> 5;
  int r0 = blockIdx.x * 16;
  int lh  = lane >> 4;               // lane half (wave32)

#ifdef HAVE_TDM
  if (threadIdx.x < 32) {            // one wave drives the Tensor Data Mover
    tdm_load_tile(agg + (size_t)r0 * HDIM, 0u);
    tdm_load_tile(xin + (size_t)r0 * HDIM, (unsigned)(16 * HDIM * 2));
    __builtin_amdgcn_s_wait_tensorcnt(0);
  }
  __syncthreads();
#else
  for (int i = threadIdx.x; i < 16 * HDIM; i += 256) {
    int r = i >> 7, c = i & 127;
    Atile[i]              = agg[(size_t)(r0 + r) * HDIM + c];
    Atile[16 * HDIM + i]  = xin[(size_t)(r0 + r) * HDIM + c];
  }
  __syncthreads();
  (void)Atile;
#endif

  v8f acc = {};
  #pragma unroll
  for (int kt = 0; kt < 8; ++kt) {
    // A fragment: 16x32 f16 layout — lanes 0-15 hold K=[kb..kb+15], lanes 16-31 K=[kb+16..kb+31].
    // Byte offset within the wave's LDS allocation (Atile sits at LDS offset 0,
    // consistent with the lds_addr fields passed to the TDM descriptors).
    unsigned aoff = (unsigned)((((kt < 4) ? 0 : 16 * HDIM)
                    + (lane & 15) * HDIM + (kt & 3) * 32 + lh * 16) * 2);
    v8h h0, h1;
    asm volatile("ds_load_b128 %0, %2\n\t"
                 "ds_load_b128 %1, %2 offset:16"
                 : "=v"(h0), "=v"(h1) : "v"(aoff));
    asm volatile("s_wait_dscnt 0" ::: "memory");
    v16h a;
    #pragma unroll
    for (int i = 0; i < 8; ++i) { a[i] = h0[i]; a[8 + i] = h1[i]; }
    v16h bf = *(const v16h*)(wswz + (size_t)(((kt * 8 + wave) * 32) + lane) * 16);
    acc = __builtin_amdgcn_wmma_f32_16x16x32_f16(
        false, a, false, bf, (short)0, acc, false, false);
  }

  int col = wave * 16 + (lane & 15);
  float bv = bias[col];
  bool has_res = (resid != nullptr);
  #pragma unroll
  for (int r = 0; r < 8; ++r) {
    int ro = r0 + r + lh * 8;        // C/D layout: VGPR r -> M=r (lanes 0-15) / M=8+r
    float v = fmaxf(acc[r] + bv, 0.0f);
    size_t o = (size_t)ro * HDIM + col;
    if (has_res) v += (float)resid[o];
    out[o] = (_Float16)v;
  }
}

// ---------------- graph pooling (400 nodes per graph, sorted batch) ----------------
__global__ void __launch_bounds__(128)
k_pool(const _Float16* __restrict__ x4, float* __restrict__ xc) {
  int b = blockIdx.x;
  int h = threadIdx.x;
  const _Float16* base = x4 + (size_t)b * 400 * HDIM + h;
  float mx = -3.0e38f, sm = 0.0f;
  for (int i = 0; i < 400; ++i) {
    float v = (float)base[(size_t)i * HDIM];
    mx = fmaxf(mx, v);
    sm += v;
  }
  xc[(size_t)b * XCW + h]        = mx;
  xc[(size_t)b * XCW + HDIM + h] = sm * (1.0f / 400.0f);
}

// ---------------- Conv1d(84->128,k=3) + relu + avgpool5 ----------------
__global__ void k_conv_pool(const float* __restrict__ tgt, const float* __restrict__ Wc,
                            const float* __restrict__ bc, float* __restrict__ xs) {
  int idx = blockIdx.x * blockDim.x + threadIdx.x;
  if (idx >= NGRAPH * HDIM * TSTEPS) return;
  int t  = idx % TSTEPS;
  int oc = (idx / TSTEPS) % HDIM;
  int b  = idx / (TSTEPS * HDIM);
  float s = 0.0f;
  #pragma unroll
  for (int p = 0; p < 5; ++p) {
    int pos = t * 5 + p;
    float acc = bc[oc];
    for (int ic = 0; ic < 84; ++ic) {
      const float* tr = tgt + ((size_t)b * 84 + ic) * 100 + pos;
      const float* wr = Wc + ((size_t)oc * 84 + ic) * 3;
      acc += tr[0] * wr[0] + tr[1] * wr[1] + tr[2] * wr[2];
    }
    s += fmaxf(acc, 0.0f);
  }
  xs[((size_t)t * NGRAPH + b) * HDIM + oc] = s * 0.2f;
}

// ---------------- GRU input projection: gi = xs @ Wih^T + bih ----------------
__global__ void k_gi(const float* __restrict__ xs, const float* __restrict__ Wih,
                     const float* __restrict__ bih, float* __restrict__ gi) {
  int idx = blockIdx.x * blockDim.x + threadIdx.x;
  if (idx >= TSTEPS * NGRAPH * 360) return;
  int j  = idx % 360;
  int tb = idx / 360;
  const float* xr = xs + (size_t)tb * HDIM;
  const float* wr = Wih + (size_t)j * HDIM;
  float acc = bih[j];
  for (int k = 0; k < HDIM; ++k) acc += xr[k] * wr[k];
  gi[idx] = acc;
}

// ---------------- GRU recurrence: one block per (batch, dir), h in LDS ----------------
__global__ void __launch_bounds__(128)
k_gru(const float* __restrict__ gif, const float* __restrict__ gib,
      const float* __restrict__ WhhF, const float* __restrict__ bhhF,
      const float* __restrict__ WhhB, const float* __restrict__ bhhB,
      float* __restrict__ xc) {
  int b   = blockIdx.x >> 1;
  int dir = blockIdx.x & 1;
  const float* gi  = dir ? gib : gif;
  const float* Whh = dir ? WhhB : WhhF;
  const float* bhh = dir ? bhhB : bhhF;
  __shared__ float h[GHID];
  int j = threadIdx.x;
  if (j < GHID) h[j] = 0.0f;
  __syncthreads();
  for (int step = 0; step < TSTEPS; ++step) {
    int t = dir ? (TSTEPS - 1 - step) : step;
    float h2 = 0.0f;
    if (j < GHID) {
      const float* g  = gi + ((size_t)t * NGRAPH + b) * 360;
      const float* wr = Whh + (size_t)j * GHID;
      const float* wz = Whh + (size_t)(GHID + j) * GHID;
      const float* wn = Whh + (size_t)(2 * GHID + j) * GHID;
      float ghr = bhh[j], ghz = bhh[GHID + j], ghn = bhh[2 * GHID + j];
      for (int k = 0; k < GHID; ++k) {
        float hk = h[k];
        ghr += wr[k] * hk;
        ghz += wz[k] * hk;
        ghn += wn[k] * hk;
      }
      float r = 1.0f / (1.0f + expf(-(g[j] + ghr)));
      float z = 1.0f / (1.0f + expf(-(g[GHID + j] + ghz)));
      float n = tanhf(g[2 * GHID + j] + r * ghn);
      h2 = (1.0f - z) * n + z * h[j];
    }
    __syncthreads();
    if (j < GHID) {
      h[j] = h2;
      xc[(size_t)b * XCW + 256 + t * 240 + dir * GHID + j] = h2;
    }
    __syncthreads();
  }
}

// ---------------- dense head ----------------
__global__ void __launch_bounds__(128)
k_dense1(const float* __restrict__ xc, const float* __restrict__ Wd1,
         const float* __restrict__ bd1, float* __restrict__ xd) {
  int b = blockIdx.x, j = threadIdx.x;
  if (j >= 102) return;
  const float* xr = xc + (size_t)b * XCW;
  float acc = bd1[j];
  for (int i = 0; i < XCW; ++i) acc += xr[i] * Wd1[(size_t)i * 102 + j];
  xd[b * 102 + j] = fmaxf(acc, 0.0f);
}

__global__ void k_head(const float* __restrict__ xd, const float* __restrict__ Wd3,
                       const float* __restrict__ bd3, float* __restrict__ out) {
  int b = blockIdx.x * blockDim.x + threadIdx.x;
  if (b >= NGRAPH) return;
  const float* xr = xd + b * 102;
  float l0 = bd3[0], l1 = bd3[1];
  for (int j = 0; j < 102; ++j) { l0 += xr[j] * Wd3[j * 2]; l1 += xr[j] * Wd3[j * 2 + 1]; }
  float m = fmaxf(l0, l1);
  float lse = m + logf(expf(l0 - m) + expf(l1 - m));
  out[b * 2]     = l0 - lse;
  out[b * 2 + 1] = l1 - lse;
}

// ---------------- launch ----------------
extern "C" void kernel_launch(void* const* d_in, const int* in_sizes, int n_in,
                              void* d_out, int out_size, void* d_ws, size_t ws_size,
                              hipStream_t stream) {
  (void)in_sizes; (void)n_in; (void)out_size; (void)ws_size;
  const float* x      = (const float*)d_in[0];
  const int*   ei     = (const int*)d_in[1];
  // d_in[2] (batch) unused: batch is repeat(arange(256), 400), structure is fixed
  const float* tgt    = (const float*)d_in[3];
  const float* W1rel  = (const float*)d_in[4];
  const float* W1root = (const float*)d_in[5];
  const float* b1     = (const float*)d_in[6];
  const float* W2rel  = (const float*)d_in[7];
  const float* W2root = (const float*)d_in[8];
  const float* b2     = (const float*)d_in[9];
  const float* W3rel  = (const float*)d_in[10];
  const float* W3root = (const float*)d_in[11];
  const float* b3     = (const float*)d_in[12];
  const float* Wc     = (const float*)d_in[13];
  const float* bc     = (const float*)d_in[14];
  const float* WihF   = (const float*)d_in[15];
  const float* WhhF   = (const float*)d_in[16];
  const float* bihF   = (const float*)d_in[17];
  const float* bhhF   = (const float*)d_in[18];
  const float* WihB   = (const float*)d_in[19];
  const float* WhhB   = (const float*)d_in[20];
  const float* bihB   = (const float*)d_in[21];
  const float* bhhB   = (const float*)d_in[22];
  const float* Wd1    = (const float*)d_in[23];
  const float* bd1    = (const float*)d_in[24];
  const float* Wd3    = (const float*)d_in[25];
  const float* bd3    = (const float*)d_in[26];
  float* out = (float*)d_out;

  char* ws = (char*)d_ws;
  size_t off = 0;
  auto carve = [&](size_t bytes) {
    char* p = ws + off;
    off = (off + bytes + 255) & ~(size_t)255;
    return p;
  };
  _Float16* agg  = (_Float16*)carve((size_t)NNODES * HDIM * 2);  // f16 scatter-add target
  float*    agg4 = (float*)   carve((size_t)NNODES * 4 * 4);
  _Float16* x1   = (_Float16*)carve((size_t)NNODES * HDIM * 2);
  _Float16* x2   = (_Float16*)carve((size_t)NNODES * HDIM * 2);  // reused for x4
  _Float16* x3   = (_Float16*)carve((size_t)NNODES * HDIM * 2);
  _Float16* swz2 = (_Float16*)carve((size_t)256 * HDIM * 2);
  _Float16* swz3 = (_Float16*)carve((size_t)256 * HDIM * 2);
  float*    xs   = (float*)   carve((size_t)TSTEPS * NGRAPH * HDIM * 4);
  float*    gif  = (float*)   carve((size_t)TSTEPS * NGRAPH * 360 * 4);
  float*    gib  = (float*)   carve((size_t)TSTEPS * NGRAPH * 360 * 4);
  float*    xc   = (float*)   carve((size_t)NGRAPH * XCW * 4);
  float*    xd   = (float*)   carve((size_t)NGRAPH * 102 * 4);

  const int E = NEDGES;
  const int aggDw = NNODES * HDIM / 2;   // f16 agg buffer in dwords

  // ---- layer 1 (F_IN=4, scalar path) ----
  k_zero_f32<<<(NNODES * 4 + 255) / 256, 256, 0, stream>>>(agg4, NNODES * 4);
  k_edge_agg4<<<(E + 255) / 256, 256, 0, stream>>>(x, ei, agg4, E);
  k_layer1<<<(NNODES * HDIM + 255) / 256, 256, 0, stream>>>(agg4, x, W1rel, W1root, b1, x1);

  // ---- weight swizzle for WMMA B fragments ----
  k_prep_wswz<<<128, 256, 0, stream>>>(W2rel, W2root, swz2);
  k_prep_wswz<<<128, 256, 0, stream>>>(W3rel, W3root, swz3);

  // ---- layer 2 ----
  k_zero_f32<<<(aggDw + 255) / 256, 256, 0, stream>>>((float*)agg, aggDw);
  k_edge_agg128<<<(E * 32 + 255) / 256, 256, 0, stream>>>(x1, ei, agg, E);
  k_gemm_gconv<<<NNODES / 16, 256, 0, stream>>>(agg, x1, swz2, b2, nullptr, x2);

  // ---- layer 3 (+ residual x1) ----
  k_zero_f32<<<(aggDw + 255) / 256, 256, 0, stream>>>((float*)agg, aggDw);
  k_edge_agg128<<<(E * 32 + 255) / 256, 256, 0, stream>>>(x2, ei, agg, E);
  k_gemm_gconv<<<NNODES / 16, 256, 0, stream>>>(agg, x2, swz3, b3, x1, x3);

  // ---- layer 4 (reuses W3; output into x2 buffer as x4) ----
  k_zero_f32<<<(aggDw + 255) / 256, 256, 0, stream>>>((float*)agg, aggDw);
  k_edge_agg128<<<(E * 32 + 255) / 256, 256, 0, stream>>>(x3, ei, agg, E);
  k_gemm_gconv<<<NNODES / 16, 256, 0, stream>>>(agg, x3, swz3, b3, nullptr, x2);

  // ---- graph pooling -> xc[:, 0:256] ----
  k_pool<<<NGRAPH, 128, 0, stream>>>(x2, xc);

  // ---- temporal branch ----
  k_conv_pool<<<(NGRAPH * HDIM * TSTEPS + 255) / 256, 256, 0, stream>>>(tgt, Wc, bc, xs);
  int ngi = TSTEPS * NGRAPH * 360;
  k_gi<<<(ngi + 255) / 256, 256, 0, stream>>>(xs, WihF, bihF, gif);
  k_gi<<<(ngi + 255) / 256, 256, 0, stream>>>(xs, WihB, bihB, gib);
  k_gru<<<NGRAPH * 2, 128, 0, stream>>>(gif, gib, WhhF, bhhF, WhhB, bhhB, xc);

  // ---- dense head ----
  k_dense1<<<NGRAPH, 128, 0, stream>>>(xc, Wd1, bd1, xd);
  k_head<<<1, 256, 0, stream>>>(xd, Wd3, bd3, out);
}
